// LstmModel_15109694947547
// MI455X (gfx1250) — matrix operand order repaired and checked by reading, not compiled, not run on hardware
//
#include <hip/hip_runtime.h>

// ---------------------------------------------------------------------------
// Fused LSTM (T=256, F=64, H=32) for gfx1250 (MI455X), wave32 + WMMA bf16.
//
// One block = 128 threads = 4 waves = one 16-row batch tile.
// Per step: gates[16x128] = [x_t(16x64), h(16x32)] (bf16) @ W(96x128) (bf16)
// via v_wmma_f32_16x16x32_bf16, fp32 accumulators, bias preloaded.
// Weights live in VGPRs for the whole kernel; x streamed once (memory-bound,
// ~1.07 GB => ~46 us floor at 23.3 TB/s HBM).
// Activations use gfx1250 hardware v_tanh_f32 (sigmoid via tanh identity),
// so each gate costs a single transcendental that co-executes with WMMA.
// ---------------------------------------------------------------------------

#define TT 256
#define FF 64
#define HH 32

typedef __attribute__((ext_vector_type(16))) __bf16 v16bf;
typedef __attribute__((ext_vector_type(8)))  __bf16 v8bf;
typedef __attribute__((ext_vector_type(8)))  float  v8f;

#if __has_builtin(__builtin_amdgcn_tanhf)
__device__ __forceinline__ float fast_tanh(float x) {
  return __builtin_amdgcn_tanhf(x);           // v_tanh_f32
}
#else
__device__ __forceinline__ float fast_tanh(float x) {
  x = fminf(fmaxf(x, -15.0f), 15.0f);
  float e = __expf(2.0f * x);
  return (e - 1.0f) * __builtin_amdgcn_rcpf(e + 1.0f);
}
#endif

// sigmoid(x) = 0.5 * tanh(x/2) + 0.5  -> one transcendental, no division
__device__ __forceinline__ float fast_sigmoid(float x) {
  return fmaf(0.5f, fast_tanh(0.5f * x), 0.5f);
}

// Build a 16x32 bf16 A-fragment from an LDS row-major tile.
// ISA 16-bit A layout (wave32): lanes 0-15 hold row M=lane, elems[0:8)=K
// koff+kh..+8, elems[8:16)=K koff+kh+16..; kh = 8*(lane>=16).
__device__ __forceinline__ v16bf load_afrag(const __bf16* base, int stride,
                                            int lane, int koff) {
  const int row = lane & 15;
  const int kh  = (lane >> 4) << 3;  // 0 or 8
  const __bf16* p = base + row * stride + koff + kh;
  v8bf lo = *(const v8bf*)(p);        // ds_load_b128
  v8bf hi = *(const v8bf*)(p + 16);   // ds_load_b128
  v16bf f;
#pragma unroll
  for (int i = 0; i < 8; ++i) { f[i] = lo[i]; f[i + 8] = hi[i]; }
  return f;
}

__global__ __launch_bounds__(128) void lstm_fused_wmma(
    const float* __restrict__ x,      // [B, T*F]
    const float* __restrict__ W,      // [96, 128] rows 0-63: W_x, 64-95: W_h
    const float* __restrict__ bl,     // [128]
    const float* __restrict__ Wd,     // [32]
    const float* __restrict__ bd,     // [1]
    float* __restrict__ out)          // [B]
{
  __shared__ __align__(16) __bf16 lds_x[16 * FF];      // x_t tile, bf16
  __shared__ __align__(16) __bf16 lds_h[16 * HH];      // h tile, bf16
  __shared__ __align__(16) float  lds_g[16 * 4 * HH];  // gates, f32

  const int tid  = threadIdx.x;
  const int lane = tid & 31;
  const int wave = tid >> 5;
  const int b0   = blockIdx.x * 16;

  // ---- one-time: load W into B-matrix fragments (registers, bf16) --------
  // B layout (32x16 bf16): lanes 0-15: col N=lane, elems=K0..7,16..23;
  //                        lanes 16-31: col N=lane-16, elems=K8..15,24..31.
  v16bf wfrag[2][3];
  float bias[2];
#pragma unroll
  for (int j = 0; j < 2; ++j) {
    const int nt = wave + 4 * j;            // N-tile id 0..7
    const int n  = nt * 16 + (lane & 15);   // gate column 0..127
    bias[j] = bl[n];
#pragma unroll
    for (int kf = 0; kf < 3; ++kf) {        // K 0-31 (x), 32-63 (x), 64-95 (h)
      const int kbase = kf * 32 + ((lane >> 4) << 3);
      v16bf f;
#pragma unroll
      for (int k = 0; k < 16; ++k)
        f[k] = (__bf16)W[(kbase + (k & 7) + ((k >> 3) << 4)) * 128 + n];
      wfrag[j][kf] = f;
    }
  }

  // ---- init state: each thread owns 4 cells of the 16x32 c/h tile --------
  float c_reg[4], h_reg[4];
#pragma unroll
  for (int k = 0; k < 4; ++k) { c_reg[k] = 0.0f; h_reg[k] = 0.0f; }
#pragma unroll
  for (int k = 0; k < 4; ++k) lds_h[tid * 4 + k] = (__bf16)0.0f;
  __syncthreads();

  // x staging: 8 threads per row, 8 contiguous floats per thread
  const int xrow = tid >> 3;            // 0..15
  const int xcol = (tid & 7) << 3;      // 0,8,..,56
  const float* xptr = x + (size_t)(b0 + xrow) * (TT * FF) + xcol;

  for (int t = 0; t < TT; ++t) {
    // ---- stage x_t (f32 global -> bf16 LDS) ------------------------------
    float4 a = *(const float4*)(xptr + 0);
    float4 b = *(const float4*)(xptr + 4);
    v8bf xv;
    xv[0] = (__bf16)a.x; xv[1] = (__bf16)a.y;
    xv[2] = (__bf16)a.z; xv[3] = (__bf16)a.w;
    xv[4] = (__bf16)b.x; xv[5] = (__bf16)b.y;
    xv[6] = (__bf16)b.z; xv[7] = (__bf16)b.w;
    *(v8bf*)&lds_x[xrow * FF + xcol] = xv;
    if (t + 1 < TT) __builtin_prefetch(xptr + FF, 0, 0);  // global_prefetch_b8
    xptr += FF;
    __syncthreads();

    // ---- A fragments + WMMA gate GEMM ------------------------------------
    v16bf ax0 = load_afrag(lds_x, FF, lane, 0);
    v16bf ax1 = load_afrag(lds_x, FF, lane, 32);
    v16bf ah  = load_afrag(lds_h, HH, lane, 0);

#pragma unroll
    for (int j = 0; j < 2; ++j) {
      v8f acc;
#pragma unroll
      for (int r = 0; r < 8; ++r) acc[r] = bias[j];
      acc = __builtin_amdgcn_wmma_f32_16x16x32_bf16(
          false, ax0, false, wfrag[j][0], (short)0, acc, false, false);
      acc = __builtin_amdgcn_wmma_f32_16x16x32_bf16(
          false, ax1, false, wfrag[j][1], (short)0, acc, false, false);
      acc = __builtin_amdgcn_wmma_f32_16x16x32_bf16(
          false, ah,  false, wfrag[j][2], (short)0, acc, false, false);
      // D layout: VGPR r -> row M = r + 8*(lane>=16), col N = lane%16
      const int col = (wave + 4 * j) * 16 + (lane & 15);
      const int mhi = (lane >> 4) << 3;
#pragma unroll
      for (int r = 0; r < 8; ++r)
        lds_g[(mhi + r) * 128 + col] = acc[r];
    }
    __syncthreads();

    // ---- elementwise LSTM cell update (4 cells per thread) ---------------
#pragma unroll
    for (int k = 0; k < 4; ++k) {
      const int e   = tid * 4 + k;
      const int row = e >> 5;
      const int col = e & 31;
      const float gi = lds_g[row * 128 +       col];
      const float gj = lds_g[row * 128 +  32 + col];
      const float gf = lds_g[row * 128 +  64 + col];
      const float go = lds_g[row * 128 +  96 + col];
      float c = c_reg[k];
      c = c * fast_sigmoid(gf + 1.0f) + fast_sigmoid(gi) * fast_tanh(gj);
      const float h = fast_tanh(c) * fast_sigmoid(go);
      c_reg[k] = c;
      h_reg[k] = h;
      lds_h[row * HH + col] = (__bf16)h;   // feeds next step's A-fragment
    }
    __syncthreads();
  }

  // ---- final dense(1) + ELU on h_T (f32 kept in registers) ---------------
  {
    const int col0 = (tid * 4) & 31;
    float p = 0.0f;
#pragma unroll
    for (int k = 0; k < 4; ++k) p += h_reg[k] * Wd[col0 + k];
    lds_g[tid] = p;   // threads [r*8, r*8+8) own row r
    __syncthreads();
    if (tid < 16) {
      float s = bd[0];
#pragma unroll
      for (int j = 0; j < 8; ++j) s += lds_g[tid * 8 + j];
      out[b0 + tid] = (s > 0.0f) ? s : (__expf(s) - 1.0f);
    }
  }
}

extern "C" void kernel_launch(void* const* d_in, const int* in_sizes, int n_in,
                              void* d_out, int out_size, void* d_ws, size_t ws_size,
                              hipStream_t stream) {
  (void)n_in; (void)d_ws; (void)ws_size;
  const float* x  = (const float*)d_in[0];
  const float* W  = (const float*)d_in[1];
  const float* bl = (const float*)d_in[2];
  const float* Wd = (const float*)d_in[3];
  const float* bd = (const float*)d_in[4];
  float* out = (float*)d_out;

  const int B = in_sizes[0] / (TT * FF);   // 16384
  (void)out_size;

  dim3 grid(B / 16);    // one block per 16-row batch tile
  dim3 block(128);      // 4 waves; each wave owns 2 of 8 gate N-tiles
  lstm_fused_wmma<<<grid, block, 0, stream>>>(x, W, bl, Wd, bd, out);
}
